// TinySelfAttention_6493990551692
// MI455X (gfx1250) — matrix operand order repaired and checked by reading, not compiled, not run on hardware
//
#include <hip/hip_runtime.h>
#include <hip/hip_bf16.h>

// ---------------------------------------------------------------------------
// TinySelfAttention on MI455X (gfx1250, wave32, WMMA + async-to-LDS)
//   B=4, S=2048, D=1024, H=16, HD=64
// Pipeline:
//   1) cvt_x     : X fp32 -> Xh f16
//   2) cvt_w     : Wq/Wk/Wv fp32 [k][n] -> Wt f16 [m][n][k] (B-frags contiguous)
//   3) qkv_gemm  : WMMA f32_16x16x32_f16 -> Qh/Kh [b,h,s,d] f16, Vt [b,h,d,s] f16
//   4) flash_attn: 64-key blocks; K/V double-buffered in LDS via
//                  global_load_async_to_lds_b128 (ASYNCcnt), WMMA scores + P*V.
// Rationale: 8 waves/WG share one (b,h); LDS staging cuts K/V L2 traffic 8x
// (~4GB -> ~0.5GB), below the WMMA compute floor. 192MB L2 keeps Qh/Kh/Vt hot.
// ---------------------------------------------------------------------------

#define SEQ    2048
#define BATCH  4
#define NHEAD  16
#define HDIM   64
#define DMODEL 1024

typedef __attribute__((ext_vector_type(16))) _Float16 v16h;
typedef __attribute__((ext_vector_type(8)))  _Float16 v8h;
typedef __attribute__((ext_vector_type(8)))  float    v8f;

union F16x16 { v16h v; v8h h[2]; };

// A-fragment (16x32 f16, row-major): lane<16 -> row=lane, K [0..16);
// lane>=16 -> row=lane-16, K [16..32). Caller passes p = row + 16*(lane>>4).
__device__ __forceinline__ v16h afrag(const _Float16* p) {
  F16x16 u;
  u.h[0] = *(const v8h*)(p);
  u.h[1] = *(const v8h*)(p + 8);
  return u.v;
}
// B-fragment (32x16 f16): lane holds one column, 16 contiguous K halves at 16*(lane>>4).
__device__ __forceinline__ v16h bfrag(const _Float16* p) {
  return *(const v16h*)p;
}

// Async 16B global->LDS copy (ASYNCcnt-tracked). LDS byte offset = low 32 bits of
// the generic address of a __shared__ object (flat LDS aperture).
__device__ __forceinline__ void async_b128(const _Float16* lds_dst, const _Float16* gsrc) {
  unsigned lds_off = (unsigned)(uintptr_t)lds_dst;
  unsigned long long ga = (unsigned long long)(uintptr_t)gsrc;
  asm volatile("global_load_async_to_lds_b128 %0, %1, off"
               :: "v"(lds_off), "v"(ga) : "memory");
}
__device__ __forceinline__ void wait_async0() {
  asm volatile("s_wait_asynccnt 0x0" ::: "memory");
}

// ---------------------------------------------------------------------------
__global__ void cvt_x(const float* __restrict__ X, _Float16* __restrict__ Xh, int n) {
  int i = blockIdx.x * blockDim.x + threadIdx.x;
  if (i < n) Xh[i] = (_Float16)X[i];
}

// Wt[m][n][k] = Wm[k][n]
__global__ void cvt_w(const float* __restrict__ Wq, const float* __restrict__ Wk,
                      const float* __restrict__ Wv, _Float16* __restrict__ Wt) {
  int i = blockIdx.x * blockDim.x + threadIdx.x;      // 0 .. 3*2^20-1
  int m = i >> 20;
  int r = i & ((1 << 20) - 1);
  int n = r >> 10;
  int k = r & 1023;
  const float* W = (m == 0) ? Wq : ((m == 1) ? Wk : Wv);
  Wt[i] = (_Float16)W[k * DMODEL + n];
}

// ---------------------------------------------------------------------------
// QKV projection: each wave computes a 16x64 tile of one of Q/K/V.
__global__ void qkv_gemm(const _Float16* __restrict__ Xh,
                         const _Float16* __restrict__ Wt,
                         const float* __restrict__ bq,
                         const float* __restrict__ bk,
                         const float* __restrict__ bv,
                         _Float16* __restrict__ Qh,
                         _Float16* __restrict__ Kh,
                         _Float16* __restrict__ Vt) {
  const int lane = threadIdx.x & 31;
  const int wid  = blockIdx.x * (blockDim.x >> 5) + (threadIdx.x >> 5);
  const int mat  = wid >> 13;          // 0=Q 1=K 2=V
  const int rem  = wid & 8191;
  const int rowt = rem >> 4;           // 512 row tiles of 16
  const int ct   = rem & 15;           // 16 col groups of 64
  const int col_base = ct * 64;
  const int half = lane >> 4;
  const int l15  = lane & 15;

  const _Float16* W    = Wt + (size_t)mat * (DMODEL * DMODEL);
  const _Float16* arow = Xh + (size_t)(rowt * 16 + l15) * DMODEL + half * 16;

  v8f acc[4] = {};
  for (int kc = 0; kc < DMODEL; kc += 32) {
    v16h a = afrag(arow + kc);
#pragma unroll
    for (int c = 0; c < 4; ++c) {
      const _Float16* bp = W + (size_t)(col_base + c * 16 + l15) * DMODEL + kc + half * 16;
      v16h b = bfrag(bp);
      acc[c] = __builtin_amdgcn_wmma_f32_16x16x32_f16(false, a, false, b,
                                                      (short)0, acc[c], false, false);
    }
  }

  const float* bias = (mat == 0) ? bq : ((mat == 1) ? bk : bv);
  const int h = col_base >> 6;
#pragma unroll
  for (int c = 0; c < 4; ++c) {
    int   n  = col_base + c * 16 + l15;
    float bb = bias[n];
    int   d  = n & 63;
#pragma unroll
    for (int v = 0; v < 8; ++v) {
      int   row = rowt * 16 + v + half * 8;   // C layout: VGPR v -> row v + 8*(lane>=16)
      int   b_  = row >> 11;
      int   s   = row & 2047;
      float val = acc[c][v] + bb;
      if (mat == 2) {
        Vt[((size_t)((b_ * NHEAD + h) * HDIM + d)) * SEQ + s] = (_Float16)val;
      } else {
        _Float16* dst = (mat == 0) ? Qh : Kh;
        dst[((size_t)((b_ * NHEAD + h) * SEQ + s)) * HDIM + d] = (_Float16)val;
      }
    }
  }
}

// ---------------------------------------------------------------------------
// Flash attention: 8 waves/block, all share (b,h); wave w handles q-tile qtg*8+w.
// 64-key blocks, K/V double-buffered in LDS via async copies.
__global__ void __launch_bounds__(256)
flash_attn(const _Float16* __restrict__ Qh,
           const _Float16* __restrict__ Kh,
           const _Float16* __restrict__ Vt,
           const float* __restrict__ mask,
           float* __restrict__ out) {
  __shared__ _Float16 KbS[2][64][HDIM];   // [buf][key][d]     8KB x2
  __shared__ _Float16 VbS[2][HDIM][64];   // [buf][d][key]     8KB x2
  __shared__ _Float16 PbS[8][16 * 64];    // per-wave P tiles  16KB

  const int tid  = threadIdx.x;
  const int lane = tid & 31;
  const int warp = tid >> 5;
  const int bh   = blockIdx.x >> 4;       // 0..63
  const int qtg  = blockIdx.x & 15;
  const int qt   = qtg * 8 + warp;        // 0..127
  const int b    = bh >> 4;
  const int h    = bh & 15;
  const int half = lane >> 4;
  const int l15  = lane & 15;

  const _Float16* Q = Qh + (size_t)(b * NHEAD + h) * SEQ * HDIM;
  const _Float16* K = Kh + (size_t)(b * NHEAD + h) * SEQ * HDIM;
  const _Float16* V = Vt + (size_t)(b * NHEAD + h) * HDIM * SEQ;  // [d][s]
  const float*  msk = mask + b * SEQ;

  // Q A-frags (inner-d 0..31 / 32..63) stay resident.
  const _Float16* qrow = Q + (size_t)(qt * 16 + l15) * HDIM + half * 16;
  v16h qa0 = afrag(qrow);
  v16h qa1 = afrag(qrow + 32);

  v8f   o[4] = {};
  float mrun[8], lrun[8];
#pragma unroll
  for (int v = 0; v < 8; ++v) { mrun[v] = -1e30f; lrun[v] = 0.0f; }

  const float scale = 0.125f;             // 1/sqrt(64)
  _Float16* pl = PbS[warp];

  // Cooperative async copy of one 64-key K/V block into buffer `buf`.
  // 256 threads x 2 chunks x 16B = 8KB each for K and V.
  auto issue_copy = [&](int buf, int kb) {
#pragma unroll
    for (int i = 0; i < 2; ++i) {
      int c   = tid + 256 * i;            // 0..511
      int row = c >> 3;                   // 64 rows (K: key, V: d)
      int off = (c & 7) * 8;              // halves (16B chunks)
      async_b128(&KbS[buf][row][off], K + (size_t)(kb + row) * HDIM + off);
      async_b128(&VbS[buf][row][off], V + (size_t)row * SEQ + kb + off);
    }
  };

  issue_copy(0, 0);
  int cur = 0;

  for (int kb = 0; kb < SEQ; kb += 64) {
    wait_async0();
    __syncthreads();                      // buf[cur] ready; buf[cur^1] drained
    if (kb + 64 < SEQ) issue_copy(cur ^ 1, kb + 64);

    // --- scores: 4 key-tiles x (2 WMMAs over inner-d 64), K B-frags from LDS
    v8f sc[4];
#pragma unroll
    for (int nt = 0; nt < 4; ++nt) {
      v16h kf0 = bfrag(&KbS[cur][nt * 16 + l15][half * 16]);
      v16h kf1 = bfrag(&KbS[cur][nt * 16 + l15][32 + half * 16]);
      v8f s = {};
      s = __builtin_amdgcn_wmma_f32_16x16x32_f16(false, qa0, false, kf0, (short)0, s, false, false);
      s = __builtin_amdgcn_wmma_f32_16x16x32_f16(false, qa1, false, kf1, (short)0, s, false, false);
      sc[nt] = s;
    }
    float mk[4];
#pragma unroll
    for (int nt = 0; nt < 4; ++nt) mk[nt] = msk[kb + nt * 16 + l15];

    // --- online softmax (row = v + 8*half; shuffles stay inside 16-lane halves)
#pragma unroll
    for (int v = 0; v < 8; ++v) {
      float x[4];
#pragma unroll
      for (int nt = 0; nt < 4; ++nt) x[nt] = sc[nt][v] * scale + mk[nt];
      float bm = fmaxf(fmaxf(x[0], x[1]), fmaxf(x[2], x[3]));
      bm = fmaxf(bm, __shfl_xor(bm, 1, 32));
      bm = fmaxf(bm, __shfl_xor(bm, 2, 32));
      bm = fmaxf(bm, __shfl_xor(bm, 4, 32));
      bm = fmaxf(bm, __shfl_xor(bm, 8, 32));
      float nm    = fmaxf(mrun[v], bm);
      float alpha = __expf(mrun[v] - nm);
      mrun[v] = nm;
      float p[4], rs = 0.0f;
#pragma unroll
      for (int nt = 0; nt < 4; ++nt) { p[nt] = __expf(x[nt] - nm); rs += p[nt]; }
      rs += __shfl_xor(rs, 1, 32);
      rs += __shfl_xor(rs, 2, 32);
      rs += __shfl_xor(rs, 4, 32);
      rs += __shfl_xor(rs, 8, 32);
      lrun[v] = lrun[v] * alpha + rs;
#pragma unroll
      for (int c = 0; c < 4; ++c) o[c][v] *= alpha;
      int r = v + half * 8;
#pragma unroll
      for (int nt = 0; nt < 4; ++nt)
        pl[r * 64 + nt * 16 + l15] = (_Float16)p[nt];
    }
    __asm__ volatile("s_wait_dscnt 0x0" ::: "memory");   // P store->load turnaround (per-wave)

    // --- P (16x64, two A-frags) * V (B-frags from LDS, contiguous along keys)
    const _Float16* pp = pl + l15 * 64 + half * 16;
    F16x16 pa0, pa1;
    pa0.h[0] = *(const v8h*)(pp);
    pa0.h[1] = *(const v8h*)(pp + 8);
    pa1.h[0] = *(const v8h*)(pp + 32);
    pa1.h[1] = *(const v8h*)(pp + 40);
#pragma unroll
    for (int c = 0; c < 4; ++c) {
      v16h vf0 = bfrag(&VbS[cur][c * 16 + l15][half * 16]);
      v16h vf1 = bfrag(&VbS[cur][c * 16 + l15][32 + half * 16]);
      o[c] = __builtin_amdgcn_wmma_f32_16x16x32_f16(false, pa0.v, false, vf0, (short)0, o[c], false, false);
      o[c] = __builtin_amdgcn_wmma_f32_16x16x32_f16(false, pa1.v, false, vf1, (short)0, o[c], false, false);
    }
    cur ^= 1;
  }

  // --- epilogue: normalize, write out[b][q][h*64+d] fp32
#pragma unroll
  for (int v = 0; v < 8; ++v) {
    int   r   = v + half * 8;
    int   q   = qt * 16 + r;
    float inv = 1.0f / lrun[v];
#pragma unroll
    for (int c = 0; c < 4; ++c) {
      out[((size_t)(b * SEQ + q)) * DMODEL + h * HDIM + c * 16 + l15] = o[c][v] * inv;
    }
  }
}

// ---------------------------------------------------------------------------
extern "C" void kernel_launch(void* const* d_in, const int* in_sizes, int n_in,
                              void* d_out, int out_size, void* d_ws, size_t ws_size,
                              hipStream_t stream) {
  const float* X    = (const float*)d_in[0];
  const float* mask = (const float*)d_in[1];
  const float* Wq   = (const float*)d_in[2];
  const float* bq   = (const float*)d_in[3];
  const float* Wk   = (const float*)d_in[4];
  const float* bk   = (const float*)d_in[5];
  const float* Wv   = (const float*)d_in[6];
  const float* bv   = (const float*)d_in[7];
  float* out = (float*)d_out;

  char* ws = (char*)d_ws;
  _Float16* Xh = (_Float16*)ws;                                   // 16 MiB
  _Float16* Wt = (_Float16*)(ws + (size_t)16777216);              //  6 MiB
  _Float16* Qh = (_Float16*)(ws + (size_t)16777216 + 6291456);    // 16 MiB
  _Float16* Kh = Qh + (size_t)8192 * 1024;                        // 16 MiB
  _Float16* Vt = Kh + (size_t)8192 * 1024;                        // 16 MiB

  cvt_x<<<(8192 * 1024) / 256, 256, 0, stream>>>(X, Xh, 8192 * 1024);
  cvt_w<<<(3 * 1024 * 1024) / 256, 256, 0, stream>>>(Wq, Wk, Wv, Wt);
  qkv_gemm<<<3072, 256, 0, stream>>>(Xh, Wt, bq, bk, bv, Qh, Kh, Vt);
  flash_attn<<<1024, 256, 0, stream>>>(Qh, Kh, Vt, mask, out);
}
